// MarkovStationaryPooling_3599182594563
// MI455X (gfx1250) — compile-verified, matrix operand order checked by reading, not run on hardware
//
#include <hip/hip_runtime.h>

// ---------------- types ----------------
typedef __attribute__((ext_vector_type(16))) __bf16 bf16x16;
typedef __attribute__((ext_vector_type(8)))  __bf16 bf16x8;
typedef __attribute__((ext_vector_type(4)))  __bf16 bf16x4;
typedef __attribute__((ext_vector_type(8)))  float  f32x8;

#define BATCH 16
#define SEQ   2048
#define DIM   1024
#define DKDIM 1024
#define SM_SCALE 0.03125f   // 1/sqrt(1024)

// wave-level register blocking: each wave owns MT x NT tiles of 16x16
#define MT 2   // 32 rows
#define NT 4   // 64 cols

union ABv { bf16x16 v; bf16x8 h[2]; };

// ---------------- conversion / transpose ----------------
__global__ void cvt_f32_to_bf16x4(const float* __restrict__ src,
                                  __bf16* __restrict__ dst, long long n4) {
  long long i = (long long)blockIdx.x * blockDim.x + threadIdx.x;
  if (i >= n4) return;
  float4 f = ((const float4*)src)[i];
  bf16x4 o = { (__bf16)f.x, (__bf16)f.y, (__bf16)f.z, (__bf16)f.w };
  ((bf16x4*)dst)[i] = o;
}

// dst[n*K + k] = src[k*N + n]  (store W^T in bf16)
__global__ void transpose_f32_to_bf16(const float* __restrict__ src,
                                      __bf16* __restrict__ dst, int K, int N) {
  long long idx = (long long)blockIdx.x * blockDim.x + threadIdx.x;
  if (idx >= (long long)K * N) return;
  int k = (int)(idx / N);
  int n = (int)(idx % N);
  dst[(size_t)n * K + k] = (__bf16)src[idx];
}

// ---------------- WMMA register-blocked GEMM core ----------------
// Computes an MT*16 x NT*16 f32 block of  C = A * Bt^T, where
// A  : Mrows x Kdim row-major bf16
// Bt : N     x Kdim row-major bf16  (i.e. B pre-transposed; for Q*K^T, Bt = K)
// Per the 16-bit A/B WMMA VGPR layout: lane&15 selects the row (A) / column (B),
// lanes 0-15 carry K {0..7,16..23}, lanes 16-31 carry K {8..15,24..31} — i.e.
// two contiguous 8 x bf16 (16B) chunks per lane, per operand, per k-step.
__device__ __forceinline__ void wmma_block(const __bf16* __restrict__ A, int lda,
                                           const __bf16* __restrict__ Bt, int ldb,
                                           int rowBase, int colBase, int Kdim,
                                           f32x8 (&acc)[MT][NT]) {
  const int lane = threadIdx.x & 31;
  const int r    = lane & 15;
  const int kgrp = (lane >> 4) * 8;
  const __bf16* arow[MT];
  const __bf16* brow[NT];
#pragma unroll
  for (int mi = 0; mi < MT; ++mi)
    arow[mi] = A + (size_t)(rowBase + mi * 16 + r) * lda + kgrp;
#pragma unroll
  for (int ni = 0; ni < NT; ++ni)
    brow[ni] = Bt + (size_t)(colBase + ni * 16 + r) * ldb + kgrp;

#pragma unroll 2
  for (int kk = 0; kk < Kdim; kk += 32) {
    ABv a[MT], b[NT];
#pragma unroll
    for (int mi = 0; mi < MT; ++mi) {
      a[mi].h[0] = *(const bf16x8*)(arow[mi] + kk);
      a[mi].h[1] = *(const bf16x8*)(arow[mi] + kk + 16);
      if ((kk & 63) == 0)                       // one cacheline-ahead prefetch / 128B
        __builtin_prefetch(arow[mi] + kk + 256, 0, 3);  // -> global_prefetch_b8
    }
#pragma unroll
    for (int ni = 0; ni < NT; ++ni) {
      b[ni].h[0] = *(const bf16x8*)(brow[ni] + kk);
      b[ni].h[1] = *(const bf16x8*)(brow[ni] + kk + 16);
      if ((kk & 63) == 0)
        __builtin_prefetch(brow[ni] + kk + 256, 0, 3);
    }
#pragma unroll
    for (int mi = 0; mi < MT; ++mi)
#pragma unroll
      for (int ni = 0; ni < NT; ++ni)
        acc[mi][ni] = __builtin_amdgcn_wmma_f32_16x16x32_bf16(
            /*neg_a=*/false, a[mi].v, /*neg_b=*/false, b[ni].v,
            /*c_mod=*/(short)0, acc[mi][ni],
            /*reuse_a=*/false, /*reuse_b=*/false);
  }
}

// ---------------- projection GEMMs ----------------
// out (bf16) = A * Wt^T + bias   (used for Q and K)
__global__ __launch_bounds__(256)
void proj_gemm_bf16(const __bf16* __restrict__ A, const __bf16* __restrict__ Wt,
                    const float* __restrict__ bias, __bf16* __restrict__ out,
                    int Mrows, int N, int Kdim) {
  int wid = blockIdx.x * (blockDim.x >> 5) + (threadIdx.x >> 5);
  int ntiles = N / (NT * 16);
  int tm = wid / ntiles, tn = wid % ntiles;
  if (tm >= Mrows / (MT * 16)) return;
  const int rowBase = tm * (MT * 16), colBase = tn * (NT * 16);
  f32x8 acc[MT][NT] = {};
  wmma_block(A, Kdim, Wt, Kdim, rowBase, colBase, Kdim, acc);
  const int lane = threadIdx.x & 31;
  const int cn = lane & 15;
  const int rb = (lane >> 4) << 3;
#pragma unroll
  for (int ni = 0; ni < NT; ++ni) {
    const int n = colBase + ni * 16 + cn;
    const float bn = bias[n];
#pragma unroll
    for (int mi = 0; mi < MT; ++mi) {
      const int m0 = rowBase + mi * 16 + rb;
#pragma unroll
      for (int rr = 0; rr < 8; ++rr)
        out[(size_t)(m0 + rr) * N + n] = (__bf16)(acc[mi][ni][rr] + bn);
    }
  }
}

// out (f32) = A * Wt^T + bias   (used for V)
__global__ __launch_bounds__(256)
void proj_gemm_f32(const __bf16* __restrict__ A, const __bf16* __restrict__ Wt,
                   const float* __restrict__ bias, float* __restrict__ out,
                   int Mrows, int N, int Kdim) {
  int wid = blockIdx.x * (blockDim.x >> 5) + (threadIdx.x >> 5);
  int ntiles = N / (NT * 16);
  int tm = wid / ntiles, tn = wid % ntiles;
  if (tm >= Mrows / (MT * 16)) return;
  const int rowBase = tm * (MT * 16), colBase = tn * (NT * 16);
  f32x8 acc[MT][NT] = {};
  wmma_block(A, Kdim, Wt, Kdim, rowBase, colBase, Kdim, acc);
  const int lane = threadIdx.x & 31;
  const int cn = lane & 15;
  const int rb = (lane >> 4) << 3;
#pragma unroll
  for (int ni = 0; ni < NT; ++ni) {
    const int n = colBase + ni * 16 + cn;
    const float bn = bias[n];
#pragma unroll
    for (int mi = 0; mi < MT; ++mi) {
      const int m0 = rowBase + mi * 16 + rb;
#pragma unroll
      for (int rr = 0; rr < 8; ++rr)
        out[(size_t)(m0 + rr) * N + n] = acc[mi][ni][rr] + bn;
    }
  }
}

// ---------------- logits: M[b] = (Q[b] * K[b]^T) / sqrt(DK) ----------------
__global__ __launch_bounds__(256)
void logits_gemm(const __bf16* __restrict__ Qb, const __bf16* __restrict__ Kb,
                 float* __restrict__ Mbuf) {
  int wid = blockIdx.x * (blockDim.x >> 5) + (threadIdx.x >> 5);
  const int tilesN = SEQ / (NT * 16);             // 32
  const int tilesM = SEQ / (MT * 16);             // 64
  const int perB   = tilesN * tilesM;             // 2048
  int b  = wid / perB;
  int t  = wid % perB;
  int tm = t / tilesN, tn = t % tilesN;
  const int rowBase = tm * (MT * 16), colBase = tn * (NT * 16);
  const __bf16* A  = Qb + (size_t)b * SEQ * DKDIM;
  const __bf16* Bt = Kb + (size_t)b * SEQ * DKDIM;
  f32x8 acc[MT][NT] = {};
  wmma_block(A, DKDIM, Bt, DKDIM, rowBase, colBase, DKDIM, acc);
  float* out = Mbuf + (size_t)b * SEQ * SEQ;
  const int lane = threadIdx.x & 31;
  const int cn = lane & 15;
  const int rb = (lane >> 4) << 3;
#pragma unroll
  for (int ni = 0; ni < NT; ++ni) {
    const int n = colBase + ni * 16 + cn;
#pragma unroll
    for (int mi = 0; mi < MT; ++mi) {
      const int m0 = rowBase + mi * 16 + rb;
#pragma unroll
      for (int rr = 0; rr < 8; ++rr)
        out[(size_t)(m0 + rr) * SEQ + n] = acc[mi][ni][rr] * SM_SCALE;
    }
  }
}

// ---------------- softmax over last dim, one block per row ----------------
__global__ __launch_bounds__(256)
void softmax_rows(float* __restrict__ Mbuf) {
  __shared__ float red[256];
  const int tid = threadIdx.x;
  float* p = Mbuf + (size_t)blockIdx.x * SEQ;
  float v[SEQ / 256];
  float mx = -3.402823466e38f;
#pragma unroll
  for (int i = 0; i < SEQ / 256; ++i) {
    v[i] = p[tid + i * 256];
    mx = fmaxf(mx, v[i]);
  }
  red[tid] = mx; __syncthreads();
  for (int s = 128; s > 0; s >>= 1) {
    if (tid < s) red[tid] = fmaxf(red[tid], red[tid + s]);
    __syncthreads();
  }
  mx = red[0]; __syncthreads();
  float sum = 0.f;
#pragma unroll
  for (int i = 0; i < SEQ / 256; ++i) {
    v[i] = __expf(v[i] - mx);
    sum += v[i];
  }
  red[tid] = sum; __syncthreads();
  for (int s = 128; s > 0; s >>= 1) {
    if (tid < s) red[tid] += red[tid + s];
    __syncthreads();
  }
  const float inv = 1.0f / red[0];
#pragma unroll
  for (int i = 0; i < SEQ / 256; ++i)
    p[tid + i * 256] = v[i] * inv;
}

// ---------------- power iteration: dst[b,m] = sum_l src[b,l]*M[b,l,m] ----------
__global__ __launch_bounds__(256)
void init_pi(float* __restrict__ pi) {
  pi[blockIdx.x * 256 + threadIdx.x] = 1.0f / (float)SEQ;
}

__global__ __launch_bounds__(256)
void power_iter(const float* __restrict__ Mbuf, const float* __restrict__ src,
                float* __restrict__ dst) {
  const int b = blockIdx.x >> 3;                 // 8 blocks per batch
  const int m = ((blockIdx.x & 7) << 8) + threadIdx.x;
  const float* Mb = Mbuf + (size_t)b * SEQ * SEQ;
  const float* pv = src + b * SEQ;
  float acc = 0.f;
#pragma unroll 8
  for (int l = 0; l < SEQ; ++l)
    acc = fmaf(pv[l], Mb[(size_t)l * SEQ + m], acc);
  dst[b * SEQ + m] = acc;
}

// ---------------- pooled output: out[b,d] = sum_l pi[b,l]*V[b,l,d] ----------
__global__ __launch_bounds__(256)
void pool_kernel(const float* __restrict__ pi, const float* __restrict__ V,
                 float* __restrict__ out) {
  const int b = blockIdx.x >> 2;                 // 4 blocks per batch
  const int d = ((blockIdx.x & 3) << 8) + threadIdx.x;
  const float* Vb = V + (size_t)b * SEQ * DIM;
  const float* pv = pi + b * SEQ;
  float acc = 0.f;
#pragma unroll 8
  for (int l = 0; l < SEQ; ++l)
    acc = fmaf(pv[l], Vb[(size_t)l * DIM + d], acc);
  out[b * DIM + d] = acc;
}

// ---------------- host launcher ----------------
extern "C" void kernel_launch(void* const* d_in, const int* in_sizes, int n_in,
                              void* d_out, int out_size, void* d_ws, size_t ws_size,
                              hipStream_t stream) {
  (void)in_sizes; (void)n_in; (void)out_size; (void)ws_size;
  const float* x  = (const float*)d_in[0];
  const float* Wq = (const float*)d_in[1];
  const float* bq = (const float*)d_in[2];
  const float* Wk = (const float*)d_in[3];
  const float* bk = (const float*)d_in[4];
  const float* Wv = (const float*)d_in[5];
  const float* bv = (const float*)d_in[6];
  float* out = (float*)d_out;

  // workspace carve-up
  char* ws = (char*)d_ws;
  size_t off = 0;
  __bf16* xb  = (__bf16*)(ws + off); off += (size_t)BATCH * SEQ * DIM * 2;     // 64 MB
  __bf16* wqt = (__bf16*)(ws + off); off += (size_t)DIM * DKDIM * 2;           // 2 MB
  __bf16* wkt = (__bf16*)(ws + off); off += (size_t)DIM * DKDIM * 2;           // 2 MB
  __bf16* wvt = (__bf16*)(ws + off); off += (size_t)DIM * DIM * 2;             // 2 MB
  __bf16* Qb  = (__bf16*)(ws + off); off += (size_t)BATCH * SEQ * DKDIM * 2;   // 64 MB
  __bf16* Kb  = (__bf16*)(ws + off); off += (size_t)BATCH * SEQ * DKDIM * 2;   // 64 MB
  float*  Vf  = (float*) (ws + off); off += (size_t)BATCH * SEQ * DIM * 4;     // 128 MB
  float*  Mb  = (float*) (ws + off); off += (size_t)BATCH * SEQ * SEQ * 4;     // 256 MB
  float*  piA = (float*) (ws + off); off += (size_t)BATCH * SEQ * 4;
  float*  piB = (float*) (ws + off); off += (size_t)BATCH * SEQ * 4;

  const int Mrows = BATCH * SEQ;                 // 32768

  // 1) convert x -> bf16 (float4/bf16x4 vectorized)
  {
    long long n4 = (long long)Mrows * DIM / 4;   // 8388608
    cvt_f32_to_bf16x4<<<(unsigned)(n4 / 256), 256, 0, stream>>>(x, xb, n4);
  }
  // 1b) transpose weights -> bf16 W^T
  {
    unsigned g = (DIM * DKDIM) / 256;            // 4096
    transpose_f32_to_bf16<<<g, 256, 0, stream>>>(Wq, wqt, DIM, DKDIM);
    transpose_f32_to_bf16<<<g, 256, 0, stream>>>(Wk, wkt, DIM, DKDIM);
    transpose_f32_to_bf16<<<g, 256, 0, stream>>>(Wv, wvt, DIM, DIM);
  }
  // 2) projections (WMMA bf16, f32 accumulate; 32x64 per wave)
  {
    unsigned waves  = (unsigned)((Mrows / (MT * 16)) * (DKDIM / (NT * 16))); // 16384
    unsigned blocks = waves / 8;                                             // 2048
    proj_gemm_bf16<<<blocks, 256, 0, stream>>>(xb, wqt, bq, Qb, Mrows, DKDIM, DIM);
    proj_gemm_bf16<<<blocks, 256, 0, stream>>>(xb, wkt, bk, Kb, Mrows, DKDIM, DIM);
    proj_gemm_f32 <<<blocks, 256, 0, stream>>>(xb, wvt, bv, Vf, Mrows, DIM,   DIM);
  }
  // 3) logits = Q K^T / sqrt(DK)  (WMMA bf16; 32x64 per wave)
  {
    unsigned waves  = (unsigned)BATCH * (SEQ / (MT * 16)) * (SEQ / (NT * 16)); // 32768
    unsigned blocks = waves / 8;                                               // 4096
    logits_gemm<<<blocks, 256, 0, stream>>>(Qb, Kb, Mb);
  }
  // 4) row softmax -> row-stochastic M
  softmax_rows<<<(unsigned)(BATCH * SEQ), 256, 0, stream>>>(Mb);
  // 5) 20 power iterations (ping-pong; even count ends back in piA)
  init_pi<<<(BATCH * SEQ) / 256, 256, 0, stream>>>(piA);
  {
    const unsigned g = BATCH * (SEQ / 256);      // 128
    float* s = piA; float* d = piB;
    for (int it = 0; it < 20; ++it) {
      power_iter<<<g, 256, 0, stream>>>(Mb, s, d);
      float* t = s; s = d; d = t;
    }
    // after 20 iterations result is in piA (== s)
  }
  // 6) pooled output
  pool_kernel<<<BATCH * (DIM / 256), 256, 0, stream>>>(piA, Vf, out);
}
